// UnifiedAttentionModule_82025285419554
// MI455X (gfx1250) — matrix operand order repaired and checked by reading, not compiled, not run on hardware
//
#include <hip/hip_runtime.h>
#include <hip/hip_bf16.h>

#define N_NODES 50000
#define E0_EDGES 800000
#define E_EDGES 850000            // E0 + N self-loops
#define IN_CH 128
#define HC 64                     // H*C
#define ED 16
#define TLEN 96
#define PCOLS 208                 // q(64) k(64) v(64) a(8) b(8)

typedef __attribute__((ext_vector_type(16))) __bf16 v16bf;
typedef __attribute__((ext_vector_type(8)))  float  v8f;
typedef __attribute__((ext_vector_type(4)))  unsigned int u32x4;

union Frag { v16bf v; u32x4 q[2]; };

__device__ __forceinline__ unsigned short f2bf(float f) {
    unsigned u = __float_as_uint(f);
    unsigned r = ((u >> 16) & 1u) + 0x7FFFu;   // round-to-nearest-even
    return (unsigned short)((u + r) >> 16);
}
// order-preserving float <-> uint encoding for atomicMax-based segment max
__device__ __forceinline__ unsigned fenc(float f) {
    unsigned b = __float_as_uint(f);
    return (b & 0x80000000u) ? ~b : (b | 0x80000000u);
}
__device__ __forceinline__ float fdec(unsigned u) {
    unsigned b = (u & 0x80000000u) ? (u & 0x7FFFFFFFu) : ~u;
    return __uint_as_float(b);
}
__device__ __forceinline__ float wred32(float v) {
#pragma unroll
    for (int m = 16; m > 0; m >>= 1) v += __shfl_xor(v, m, 32);
    return v;
}

// ---- pack Wq|Wk|Wv|compW_a|compW_b into bf16 [208 x 128] ----
__global__ void pack_w_kernel(const float* __restrict__ Wq, const float* __restrict__ Wk,
                              const float* __restrict__ Wv, const float* __restrict__ compW,
                              unsigned short* __restrict__ wp) {
    int i = blockIdx.x * blockDim.x + threadIdx.x;
    if (i >= PCOLS * IN_CH) return;
    int col = i / IN_CH, kk = i - col * IN_CH;
    float v;
    if      (col < 64)  v = Wq[col * IN_CH + kk];
    else if (col < 128) v = Wk[(col - 64) * IN_CH + kk];
    else if (col < 192) v = Wv[(col - 128) * IN_CH + kk];
    else if (col < 200) v = compW[(col - 192) * (2 * IN_CH) + kk];
    else                v = compW[(col - 200) * (2 * IN_CH) + IN_CH + kk];
    wp[i] = f2bf(v);
}

// ---- x -> bf16 ----
__global__ void conv_x_kernel(const float* __restrict__ x, unsigned short* __restrict__ xb) {
    int i = blockIdx.x * blockDim.x + threadIdx.x;
    if (i >= N_NODES * IN_CH) return;
    xb[i] = f2bf(x[i]);
}

// ---- P[N,208] = x @ Wall^T via v_wmma_f32_16x16x32_bf16; one wave per 16x16 tile ----
__global__ void proj_wmma_kernel(const unsigned short* __restrict__ xb,
                                 const unsigned short* __restrict__ wp,
                                 float* __restrict__ P) {
    int wave = threadIdx.x >> 5;
    int lane = threadIdx.x & 31;
    int mtile = blockIdx.x * 8 + wave;
    if (mtile >= (N_NODES / 16)) return;           // wave-uniform: EXEC stays all-1s
    int nbase = blockIdx.y * 16;
    int mbase = mtile * 16;
    int half = lane >> 4;                          // 0: K low-half lanes, 1: high
    int lid  = lane & 15;

    v8f c = {0.f, 0.f, 0.f, 0.f, 0.f, 0.f, 0.f, 0.f};
#pragma unroll
    for (int ks = 0; ks < 4; ++ks) {
        // A fragment: 16x32 bf16 (row M=lid, 16 contiguous K per lane half)
        Frag a;
        const unsigned short* arow = xb + (size_t)(mbase + lid) * IN_CH + ks * 32 + half * 16;
        a.q[0] = *(const u32x4*)(arow);            // K base..base+7   -> elems 0..7
        a.q[1] = *(const u32x4*)(arow + 8);        // K base+8..+15    -> elems 8..15
        // B fragment: 32x16 bf16 (col N=lid, 16 contiguous K per lane half)
        Frag b;
        const unsigned short* brow = wp + (size_t)(nbase + lid) * IN_CH + ks * 32 + half * 16;
        b.q[0] = *(const u32x4*)(brow);
        b.q[1] = *(const u32x4*)(brow + 8);
        c = __builtin_amdgcn_wmma_f32_16x16x32_bf16(false, a.v, false, b.v,
                                                    (short)0, c, false, false);
    }
#pragma unroll
    for (int r = 0; r < 8; ++r) {
        int row = mbase + half * 8 + r;            // C/D layout: lanes 0-15 M=r, 16-31 M=8+r
        P[(size_t)row * PCOLS + nbase + lid] = c[r];
    }
}

// ---- temporal profile normalization (unbiased std), wave per node ----
__global__ void tpn_kernel(const float* __restrict__ tp, float* __restrict__ tpn) {
    int wave = threadIdx.x >> 5, lane = threadIdx.x & 31;
    int n = blockIdx.x * 8 + wave;
    if (n >= N_NODES) return;
    const float* row = tp + (size_t)n * TLEN;
    float v0 = row[lane], v1 = row[lane + 32], v2 = row[lane + 64];
    float mean = wred32(v0 + v1 + v2) * (1.f / 96.f);
    float d0 = v0 - mean, d1 = v1 - mean, d2 = v2 - mean;
    float ss = wred32(d0 * d0 + d1 * d1 + d2 * d2);
    float inv = 1.f / (sqrtf(ss * (1.f / 95.f)) + 1e-8f);
    float* o = tpn + (size_t)n * TLEN;
    o[lane] = d0 * inv; o[lane + 32] = d1 * inv; o[lane + 64] = d2 * inv;
}

// ---- ke[n,h,d] = sum_c We[h*8+c,d] * k[n,h,c]  (fuses edge_attr GEMM into node-side) ----
__global__ void ke_kernel(const float* __restrict__ P, const float* __restrict__ We,
                          float* __restrict__ ke) {
    __shared__ float sWe[HC * ED];
    int n = blockIdx.x, tid = threadIdx.x;         // 128 threads
    for (int i = tid; i < HC * ED; i += 128) sWe[i] = We[i];
    __syncthreads();
    int h = tid >> 4, d = tid & 15;
    const float* kr = P + (size_t)n * PCOLS + 64 + h * 8;
    float acc = 0.f;
#pragma unroll
    for (int cc = 0; cc < 8; ++cc) acc += sWe[(h * 8 + cc) * ED + d] * kr[cc];
    ke[(size_t)n * 128 + tid] = acc;
}

// ---- edge pass 1: alpha logits (+comp term) + atomic segment max; wave per edge ----
__global__ void edge_p1_kernel(const long long* __restrict__ EI, const float* __restrict__ EA,
                               const float* __restrict__ tpn, const float* __restrict__ P,
                               const float* __restrict__ KE, const float* __restrict__ compB,
                               float* __restrict__ alphaBuf, unsigned* __restrict__ amaxU) {
    int wave = threadIdx.x >> 5, lane = threadIdx.x & 31;
    long long e = (long long)blockIdx.x * 8 + wave;
    if (e >= E_EDGES) return;
    int src, dst;
    if (e < E0_EDGES) { src = (int)EI[e]; dst = (int)EI[E0_EDGES + e]; }
    else              { src = dst = (int)(e - E0_EDGES); }

    // temporal correlation
    const float* ts = tpn + (size_t)src * TLEN;
    const float* td = tpn + (size_t)dst * TLEN;
    float s = ts[lane] * td[lane] + ts[lane + 32] * td[lane + 32] + ts[lane + 64] * td[lane + 64];
    float corr = wred32(s) * (1.f / 96.f);
    // learned complementarity
    float lh = 0.f;
    if (lane < 8)
        lh = tanhf(P[(size_t)src * PCOLS + 192 + lane] +
                   P[(size_t)dst * PCOLS + 200 + lane] + compB[lane]);
    float learned = wred32(lh) * (1.f / 8.f);
    float comp = 0.5f * corr + 0.5f * learned;

    // per-head logits: 4 lanes per head
    int h = lane >> 2, j = lane & 3;
    const float* qr = P + (size_t)dst * PCOLS + h * 8;
    const float* kr = P + (size_t)src * PCOLS + 64 + h * 8;
    float r = (qr[j] * kr[j] + qr[j + 4] * kr[j + 4]) * 0.35355339059327f; // 1/sqrt(8)
    const float* ker = KE + (size_t)src * 128 + h * 16;
#pragma unroll
    for (int dd = j; dd < 16; dd += 4) {
        float ea = (e < E0_EDGES) ? EA[e * ED + dd] : 1.0f;  // self-loop attr = 1.0
        r += ea * ker[dd];
    }
    r += __shfl_xor(r, 1, 32);
    r += __shfl_xor(r, 2, 32);
    if (j == 0) {
        float a = r - 0.5f * comp;                 // + CW * (-comp), CW = 0.5
        a = (a >= 0.f) ? a : 0.2f * a;             // leaky relu
        alphaBuf[e * 8 + h] = a;
        atomicMax(&amaxU[(size_t)dst * 8 + h], fenc(a));
    }
}

// ---- edge pass 2: exp(alpha - max) + atomic denom; thread per (e,h) ----
__global__ void edge_p2_kernel(const long long* __restrict__ EI, float* __restrict__ alphaBuf,
                               const unsigned* __restrict__ amaxU, float* __restrict__ denom) {
    long long idx = (long long)blockIdx.x * blockDim.x + threadIdx.x;
    if (idx >= (long long)E_EDGES * 8) return;
    long long e = idx >> 3; int h = (int)(idx & 7);
    int dst = (e < E0_EDGES) ? (int)EI[E0_EDGES + e] : (int)(e - E0_EDGES);
    float ex = expf(alphaBuf[idx] - fdec(amaxU[(size_t)dst * 8 + h]));
    alphaBuf[idx] = ex;
    atomicAdd(&denom[(size_t)dst * 8 + h], ex);
}

// ---- edge pass 3: att-weighted scatter-add of v; wave per edge ----
__global__ void edge_p3_kernel(const long long* __restrict__ EI, const float* __restrict__ alphaBuf,
                               const float* __restrict__ denom, const float* __restrict__ P,
                               float* __restrict__ agg) {
    int wave = threadIdx.x >> 5, lane = threadIdx.x & 31;
    long long e = (long long)blockIdx.x * 8 + wave;
    if (e >= E_EDGES) return;
    int src, dst;
    if (e < E0_EDGES) { src = (int)EI[e]; dst = (int)EI[E0_EDGES + e]; }
    else              { src = dst = (int)(e - E0_EDGES); }
#pragma unroll
    for (int it = 0; it < 2; ++it) {
        int idx = lane + it * 32;                  // 64 = H*C channels
        int h = idx >> 3;
        float att = alphaBuf[e * 8 + h] / (denom[(size_t)dst * 8 + h] + 1e-16f);
        atomicAdd(&agg[(size_t)dst * 64 + idx], P[(size_t)src * PCOLS + 128 + idx] * att);
    }
}

// ---- final: out = agg @ Wo^T + bo ----
__global__ void out_kernel(const float* __restrict__ agg, const float* __restrict__ Wo,
                           const float* __restrict__ bo, float* __restrict__ out) {
    int gid = blockIdx.x * blockDim.x + threadIdx.x;
    if (gid >= N_NODES * 8) return;
    int n = gid >> 3, cc = gid & 7;
    const float* ar = agg + (size_t)n * 64;
    const float* wr = Wo + cc * 64;
    float acc = bo[cc];
#pragma unroll
    for (int k2 = 0; k2 < 64; ++k2) acc += ar[k2] * wr[k2];
    out[gid] = acc;
}

extern "C" void kernel_launch(void* const* d_in, const int* in_sizes, int n_in,
                              void* d_out, int out_size, void* d_ws, size_t ws_size,
                              hipStream_t stream) {
    const float*     x     = (const float*)d_in[0];
    const long long* EI    = (const long long*)d_in[1];   // int64 edge_index [2, E0]
    const float*     EA    = (const float*)d_in[2];
    const float*     tp    = (const float*)d_in[3];
    const float*     Wq    = (const float*)d_in[4];
    const float*     Wk    = (const float*)d_in[5];
    const float*     Wv    = (const float*)d_in[6];
    const float*     We    = (const float*)d_in[7];
    const float*     compW = (const float*)d_in[8];
    const float*     compB = (const float*)d_in[9];
    const float*     Wo    = (const float*)d_in[10];
    const float*     bo    = (const float*)d_in[11];
    float* out = (float*)d_out;

    // workspace layout (all region sizes are multiples of 256B)
    char* ws = (char*)d_ws;
    size_t off = 0;
    auto al = [&](size_t b) { size_t o = off; off += (b + 255) & ~(size_t)255; return o; };
    unsigned short* xb    = (unsigned short*)(ws + al((size_t)N_NODES * IN_CH * 2));
    unsigned short* wp    = (unsigned short*)(ws + al((size_t)PCOLS * IN_CH * 2));
    float*          P     = (float*)(ws + al((size_t)N_NODES * PCOLS * 4));
    float*          tpn   = (float*)(ws + al((size_t)N_NODES * TLEN * 4));
    float*          ke    = (float*)(ws + al((size_t)N_NODES * 128 * 4));
    float*          alphaBuf = (float*)(ws + al((size_t)E_EDGES * 8 * 4));
    size_t zero_off = off;
    unsigned*       amaxU = (unsigned*)(ws + al((size_t)N_NODES * 8 * 4));
    float*          denom = (float*)(ws + al((size_t)N_NODES * 8 * 4));
    float*          agg   = (float*)(ws + al((size_t)N_NODES * 64 * 4));
    size_t zero_bytes = off - zero_off;

    // zero accumulators (amaxU zero == encoded -inf sentinel; every node has a self-loop)
    hipMemsetAsync(ws + zero_off, 0, zero_bytes, stream);

    pack_w_kernel<<<(PCOLS * IN_CH + 255) / 256, 256, 0, stream>>>(Wq, Wk, Wv, compW, wp);
    conv_x_kernel<<<(N_NODES * IN_CH + 255) / 256, 256, 0, stream>>>(x, xb);

    dim3 pg((N_NODES / 16 + 7) / 8, PCOLS / 16);   // 391 x 13, 8 waves/block
    proj_wmma_kernel<<<pg, 256, 0, stream>>>(xb, wp, P);

    tpn_kernel<<<(N_NODES + 7) / 8, 256, 0, stream>>>(tp, tpn);
    ke_kernel<<<N_NODES, 128, 0, stream>>>(P, We, ke);

    int eblocks = (E_EDGES + 7) / 8;
    edge_p1_kernel<<<eblocks, 256, 0, stream>>>(EI, EA, tpn, P, ke, compB, alphaBuf, amaxU);
    edge_p2_kernel<<<(int)(((long long)E_EDGES * 8 + 255) / 256), 256, 0, stream>>>(EI, alphaBuf, amaxU, denom);
    edge_p3_kernel<<<eblocks, 256, 0, stream>>>(EI, alphaBuf, denom, P, agg);

    out_kernel<<<(N_NODES * 8 + 255) / 256, 256, 0, stream>>>(agg, Wo, bo, out);
}
